// _RPN_FPN_77610059039163
// MI455X (gfx1250) — matrix operand order repaired and checked by reading, not compiled, not run on hardware
//
#include <hip/hip_runtime.h>
#include <hip/hip_bf16.h>
#include <hip/hip_fp16.h>

typedef __attribute__((ext_vector_type(16))) _Float16 v16h;
typedef __attribute__((ext_vector_type(8)))  float    v8f;
typedef int v4i_ __attribute__((vector_size(16)));   // matches builtin's V4i param

#define CIN   256
#define COUT  512
#define KTOT  2304            // 256*9
#define NKCH  72              // 2304/32
#define NTOT  150111          // total anchors per batch across levels
#define OUT_PROB_OFF  1200888 // 4*NTOT*2
#define OUT_BBOX_OFF  2401776 // 2*4*NTOT*2

// LDS layout (all fragment loads 16B-aligned; strides chosen to spread banks)
#define LDSA_STRIDE  40       // halves per A row (80 B)
#define LDSB_STRIDE  40       // halves per B row (80 B)
#define CONV1_STRIDE 552      // halves per conv1 row (1104 B)
#define SMEM_BYTES (COUT*LDSA_STRIDE*2 + 32*LDSB_STRIDE*2)  // 40960 + 2560 = 43520

// ---- gfx1250 async global->LDS staging (ASYNCcnt-tracked), with safe fallback
#if defined(__has_builtin)
#if __has_builtin(__builtin_amdgcn_global_load_async_to_lds_b128)
#define HAVE_ASYNC_COPY 1
#endif
#endif

// typed address-space pointer casts for the async builtin
#define ASYNC_GPTR(p) ((__attribute__((address_space(1))) v4i_*)(p))
#define ASYNC_LPTR(p) ((__attribute__((address_space(3))) v4i_*)(p))

__device__ __forceinline__ void wait_async_all() {
#if defined(__has_builtin)
#if __has_builtin(__builtin_amdgcn_s_wait_asynccnt)
  __builtin_amdgcn_s_wait_asynccnt(0);
  return;
#else
  asm volatile("s_wait_asynccnt 0x0" ::: "memory");
  return;
#endif
#else
  asm volatile("s_wait_asynccnt 0x0" ::: "memory");
#endif
}

__device__ __forceinline__ v16h load_v16h(const _Float16* p0, const _Float16* p1) {
  union U { uint4 q[2]; v16h v; } u;
  u.q[0] = *(const uint4*)p0;
  u.q[1] = *(const uint4*)p1;
  return u.v;
}

// ---- prep: repack conv weights [512,256,3,3] f32 -> [chunk=72][m=512][kc=32] f16
__global__ void pack_wconv(const float* __restrict__ Wc, _Float16* __restrict__ Wpack) {
  int i = blockIdx.x * 256 + threadIdx.x;
  if (i >= COUT * KTOT) return;
  int kc    = i & 31;
  int m     = (i >> 5) & 511;
  int chunk = i >> 14;
  int K  = chunk * 32 + kc;       // K order: (kh,kw) major, ic minor
  int ic = K & 255;
  int khw = K >> 8;
  int kh = khw / 3, kw = khw - kh * 3;
  Wpack[i] = (_Float16)Wc[((m * CIN + ic) * 3 + kh) * 3 + kw];
}

// ---- prep: fuse W_cls(6x512) + W_box(12x512) -> padded [32][512] f16
__global__ void pack_whead(const float* __restrict__ Wcls, const float* __restrict__ Wbox,
                           _Float16* __restrict__ Whead) {
  int i = blockIdx.x * 256 + threadIdx.x;
  if (i >= 32 * COUT) return;
  int k = i & 511, row = i >> 9;
  float v = 0.f;
  if (row < 6)        v = Wcls[row * COUT + k];
  else if (row < 18)  v = Wbox[(row - 6) * COUT + k];
  Whead[i] = (_Float16)v;
}

// ---- main fused kernel: one level, one WG = 32 positions x 512 channels
__global__ __launch_bounds__(256)
void rpn_level_kernel(const float* __restrict__ X,
                      const _Float16* __restrict__ Wpack,
                      const _Float16* __restrict__ Whead,
                      const float* __restrict__ b_conv,
                      const float* __restrict__ b_cls,
                      const float* __restrict__ b_box,
                      float* __restrict__ out,
                      int H, int W, int anchor_base)
{
  __shared__ __align__(16) unsigned char smem[SMEM_BYTES];
  _Float16* ldsA  = (_Float16*)smem;                           // 512 x 32 f16 (padded)
  _Float16* conv1 = (_Float16*)smem;                           // aliases ldsA post K-loop
  _Float16* ldsB  = (_Float16*)(smem + COUT * LDSA_STRIDE * 2);// 32 x 32 f16 (padded)

  const int tid   = threadIdx.x;
  const int wave  = tid >> 5;
  const int lane  = tid & 31;
  const int half_ = lane >> 4;
  const int l16   = lane & 15;

  const int HW   = H * W;
  const int PTOT = 4 * HW;
  const int tileBase = blockIdx.x * 32;

  // im2col gather coordinates for this thread (lane-fast over positions)
  const int gposl = tid & 31;
  const int gic   = tid >> 5;            // 0..7, x4 strided ic per thread
  const int gp    = tileBase + gposl;
  int gb, gh, gw;
  { int p = (gp < PTOT) ? gp : 0;
    gb = p / HW; int hw = p - gb * HW; gh = hw / W; gw = hw - gh * W; }
  const float* Xb = X + (size_t)gb * CIN * (size_t)HW;

  v8f acc[4][2] = {};   // [m-subtile][n-tile], 64x32 per wave

  // incremental K-chunk decode (avoids div/mod in the hot loop)
  int kh = 0, kw = 0, sub = 0;   // khw = kh*3+kw advances every 8 chunks

  for (int kc = 0; kc < NKCH; ++kc) {
    __syncthreads();
    // ---- stage A chunk (32 KB): async global->LDS when available ----
#ifdef HAVE_ASYNC_COPY
    {
      const char* src = (const char*)(Wpack + (size_t)kc * COUT * 32);
      for (int e = tid; e < (COUT * 32) / 8; e += 256) {
        int row = e >> 2, col = e & 3;
        __builtin_amdgcn_global_load_async_to_lds_b128(
            ASYNC_GPTR(src + (size_t)e * 16),
            ASYNC_LPTR(ldsA + row * LDSA_STRIDE + col * 8),
            0, 0);
      }
    }
#else
    {
      const uint4* src = (const uint4*)(Wpack + (size_t)kc * COUT * 32);
      for (int e = tid; e < (COUT * 32) / 8; e += 256) {
        int row = e >> 2, col = e & 3;
        *(uint4*)(ldsA + row * LDSA_STRIDE + col * 8) = src[e];
      }
    }
#endif
    // ---- stage B chunk (32 pos x 32 ic halves): im2col gather w/ zero pad ----
    {
      const int ic0 = sub * 32;
      const int y = gh + kh - 1, x = gw + kw - 1;
      const bool inb = (gp < PTOT) && ((unsigned)y < (unsigned)H) && ((unsigned)x < (unsigned)W);
      const float* src = Xb + ((size_t)ic0 * (size_t)H + (size_t)(inb ? y : 0)) * (size_t)W
                            + (size_t)(inb ? x : 0);
      #pragma unroll
      for (int j = 0; j < 4; ++j) {
        int icl = gic + 8 * j;
        float v = inb ? src[(size_t)icl * HW] : 0.f;
        ldsB[gposl * LDSB_STRIDE + icl] = (_Float16)v;
      }
    }
    if (++sub == 8) { sub = 0; if (++kw == 3) { kw = 0; ++kh; } }
    if (kc + 1 < NKCH)
      __builtin_prefetch(Wpack + (size_t)(kc + 1) * COUT * 32, 0, 0);
#ifdef HAVE_ASYNC_COPY
    wait_async_all();
#endif
    __syncthreads();

    // B fragments (K=32 x N=16 each): lanes 0-15: K[0..7],K[16..23]; lanes 16-31: K[8..15],K[24..31]
    v16h bf[2];
    #pragma unroll
    for (int ni = 0; ni < 2; ++ni) {
      const _Float16* bp = ldsB + (ni * 16 + l16) * LDSB_STRIDE + half_ * 8;
      bf[ni] = load_v16h(bp, bp + 16);
    }
    // A fragments (M=16 x K=32): lanes 0-15: K[0..7],K[16..23]; lanes 16-31: K[8..15],K[24..31]
    #pragma unroll
    for (int ms = 0; ms < 4; ++ms) {
      const _Float16* ap = ldsA + (wave * 64 + ms * 16 + l16) * LDSA_STRIDE + half_ * 8;
      v16h af = load_v16h(ap, ap + 16);
      acc[ms][0] = __builtin_amdgcn_wmma_f32_16x16x32_f16(false, af, false, bf[0],
                                                          (short)0, acc[ms][0], false, false);
      acc[ms][1] = __builtin_amdgcn_wmma_f32_16x16x32_f16(false, af, false, bf[1],
                                                          (short)0, acc[ms][1], false, false);
    }
  }

  __syncthreads();
  // conv1 epilogue: bias + ReLU -> f16 [pos][cout] in LDS (aliases ldsA)
  #pragma unroll
  for (int ms = 0; ms < 4; ++ms) {
    #pragma unroll
    for (int r = 0; r < 8; ++r) {
      const int m = wave * 64 + ms * 16 + 8 * half_ + r;
      const float bias = b_conv[m];
      #pragma unroll
      for (int ni = 0; ni < 2; ++ni) {
        const int n = ni * 16 + l16;
        float v = acc[ms][ni][r] + bias;
        v = v > 0.f ? v : 0.f;
        conv1[n * CONV1_STRIDE + m] = (_Float16)v;
      }
    }
  }
  __syncthreads();

  // head GEMM: 32(head rows) x 32(pos) x K=512, waves 0..3 own one 16x16 tile each
  if (wave < 4) {
    const int mi = wave >> 1, ni = wave & 1;
    v8f hc = {};
    #pragma unroll
    for (int kk = 0; kk < COUT; kk += 32) {
      const _Float16* ap = Whead + (size_t)(mi * 16 + l16) * COUT + kk + half_ * 8;
      v16h af = load_v16h(ap, ap + 16);
      const _Float16* bp = conv1 + (ni * 16 + l16) * CONV1_STRIDE + kk + half_ * 8;
      v16h bfr = load_v16h(bp, bp + 16);
      hc = __builtin_amdgcn_wmma_f32_16x16x32_f16(false, af, false, bfr,
                                                  (short)0, hc, false, false);
    }
    // output epilogue: lane holds column = one spatial position, 8 head rows
    const int nloc = ni * 16 + l16;
    const int p = tileBase + nloc;
    if (p < PTOT) {
      const int b  = p / HW;
      const int hw = p - b * HW;
      const long nrow0 = (long)b * NTOT + anchor_base + (long)hw * 3;
      float* out_score = out;
      float* out_prob  = out + OUT_PROB_OFF;
      float* out_bbox  = out + OUT_BBOX_OFF;
      const int row0 = mi * 16 + 8 * half_;
      if (row0 == 0) {
        // rows 0..7 = score ch 0..5 + bbox ch 0..1
        float s[6], pr[6];
        #pragma unroll
        for (int c = 0; c < 6; ++c) s[c] = hc[c] + b_cls[c];
        #pragma unroll
        for (int c = 0; c < 6; ++c) {
          int partner = (c < 3) ? c + 3 : c - 3;       // softmax over (c, c+-3)
          pr[c] = 1.f / (1.f + __expf(s[partner] - s[c]));
        }
        #pragma unroll
        for (int a = 0; a < 3; ++a)
          #pragma unroll
          for (int j = 0; j < 2; ++j) {
            out_score[(nrow0 + a) * 2 + j] = s[2 * a + j];
            out_prob [(nrow0 + a) * 2 + j] = pr[2 * a + j];
          }
        out_bbox[nrow0 * 4 + 0] = hc[6] + b_box[0];
        out_bbox[nrow0 * 4 + 1] = hc[7] + b_box[1];
      } else if (row0 == 8) {
        // rows 8..15 = bbox ch 2..9
        #pragma unroll
        for (int r = 0; r < 8; ++r) {
          int cb = r + 2;
          out_bbox[(nrow0 + (cb >> 2)) * 4 + (cb & 3)] = hc[r] + b_box[cb];
        }
      } else if (row0 == 16) {
        // rows 16,17 = bbox ch 10,11
        out_bbox[(nrow0 + 2) * 4 + 2] = hc[0] + b_box[10];
        out_bbox[(nrow0 + 2) * 4 + 3] = hc[1] + b_box[11];
      } // row0 == 24: padding rows, nothing
    }
  }
}

extern "C" void kernel_launch(void* const* d_in, const int* in_sizes, int n_in,
                              void* d_out, int out_size, void* d_ws, size_t ws_size,
                              hipStream_t stream) {
  (void)in_sizes; (void)n_in; (void)out_size; (void)ws_size;
  const float* feats[5] = { (const float*)d_in[0], (const float*)d_in[1],
                            (const float*)d_in[2], (const float*)d_in[3],
                            (const float*)d_in[4] };
  // d_in[5] = im_info (unused by the computation, matching the reference)
  const float* W_conv = (const float*)d_in[6];
  const float* b_conv = (const float*)d_in[7];
  const float* W_cls  = (const float*)d_in[8];
  const float* b_cls  = (const float*)d_in[9];
  const float* W_box  = (const float*)d_in[10];
  const float* b_box  = (const float*)d_in[11];
  float* out = (float*)d_out;

  _Float16* Wpack = (_Float16*)d_ws;                               // 512*2304 f16 = 2.25 MB
  _Float16* Whead = (_Float16*)((char*)d_ws + (size_t)COUT * KTOT * 2); // 32*512 f16

  pack_wconv<<<(COUT * KTOT + 255) / 256, 256, 0, stream>>>(W_conv, Wpack);
  pack_whead<<<(32 * COUT + 255) / 256, 256, 0, stream>>>(W_cls, W_box, Whead);

  const int Hs[5]    = {150, 75, 38, 19, 10};
  const int Ws_[5]   = {250, 125, 63, 32, 16};
  const int bases[5] = {0, 112500, 140625, 147807, 149631};
  for (int l = 0; l < 5; ++l) {
    int PT = 4 * Hs[l] * Ws_[l];
    int blocks = (PT + 31) / 32;
    rpn_level_kernel<<<blocks, 256, 0, stream>>>(feats[l], Wpack, Whead,
                                                 b_conv, b_cls, b_box, out,
                                                 Hs[l], Ws_[l], bases[l]);
  }
}
